// GQA_72258529788204
// MI455X (gfx1250) — matrix operand order repaired and checked by reading, not compiled, not run on hardware
//
#include <hip/hip_runtime.h>
#include <hip/hip_bf16.h>

// ---------------------------------------------------------------------------
// Problem constants (B=2, N=2048, D=2048, QH=16, KVH=4, HD=128)
// ---------------------------------------------------------------------------
static constexpr int CB   = 2;
static constexpr int CN   = 2048;
static constexpr int CD   = 2048;
static constexpr int CQH  = 16;
static constexpr int CKVH = 4;
static constexpr int CHD  = 128;
static constexpr int CW3  = 3072;   // concat width: 2048 (q) + 512 (k) + 512 (v)

typedef __bf16 bf;
typedef __attribute__((ext_vector_type(16))) __bf16    v16bf;
typedef __attribute__((ext_vector_type(8)))  float     v8f;
typedef __attribute__((ext_vector_type(4)))  unsigned  u32x4;
typedef __attribute__((ext_vector_type(8)))  int       i32x8;
typedef __attribute__((ext_vector_type(4)))  int       i32x4;

union Frag16 { v16bf v; unsigned u[8]; };

// K index inside a 16x32 bf16 A-fragment for dword pair p (0..7), lane-half hi.
// Per ISA 7.12.2: VGPR 0..3 -> K {0..7}+hi*8, VGPR 4..7 -> K {16..23}+hi*8.
__device__ __forceinline__ int a_frag_k(int p, int hi) {
    return (p < 4 ? 2 * p : 16 + 2 * (p - 4)) + hi * 8;
}

// ---------------------------------------------------------------------------
// Tensor Data Mover support (guarded; fallback = batched uint4 -> ds_store_b128)
// ---------------------------------------------------------------------------
#if defined(__has_builtin)
#  if __has_builtin(__builtin_amdgcn_tensor_load_to_lds) && \
      __has_builtin(__builtin_amdgcn_s_wait_tensorcnt)
#    define USE_TDM 1
#  endif
#endif
#ifndef USE_TDM
#  define USE_TDM 0
#endif

#if USE_TDM
__device__ __forceinline__ unsigned lds_byte_offset(const void* p) {
    // generic -> LDS addrspace cast; ptrtoint of an AS(3) pointer is the LDS offset
    return (unsigned)(size_t)(__attribute__((address_space(3))) const void*)p;
}

// DMA a 128-row x 32-element bf16 tile (row stride = rowStride elements) to LDS.
// D# per CDNA5 ISA: group0 {count=1, lds_addr, global_addr,
// type=2}, group1 {data_size=2B, tensor_dim0=rowStride, tensor_dim1=big,
// tile_dim0=32, tile_dim1=128, tensor_dim0_stride=rowStride}.
__device__ __forceinline__ void tdm_load_tile_128x32(const bf* gsrc, unsigned lds_off,
                                                     unsigned rowStride) {
    unsigned long long ga = (unsigned long long)(size_t)gsrc;
    u32x4 g0;
    g0[0] = 1u;                                            // count=1 (valid user D#)
    g0[1] = lds_off;                                       // lds_addr
    g0[2] = (unsigned)ga;                                  // global_addr[31:0]
    g0[3] = (unsigned)((ga >> 32) & 0x01ffffffu) | (2u << 30);  // addr[56:32] | type=2
    i32x8 g1;
    g1[0] = 1 << 16;                                       // data_size=1 (2 bytes)
    g1[1] = (int)((rowStride & 0xffffu) << 16);            // tensor_dim0 lo16 @ bit48
    g1[2] = (int)(((rowStride >> 16) & 0xffffu)            // tensor_dim0 hi16
                  | (0x4000u << 16));                      // tensor_dim1 lo16 (16384 rows)
    g1[3] = (int)(32u << 16);                              // tensor_dim1 hi | tile_dim0=32
    g1[4] = 128;                                           // tile_dim1=128, tile_dim2=0
    g1[5] = (int)rowStride;                                // tensor_dim0_stride lo32
    g1[6] = 0;                                             // stride hi | dim1_stride lo
    g1[7] = 0;
    i32x4 z4; z4[0] = 0; z4[1] = 0; z4[2] = 0; z4[3] = 0;
#if __clang_major__ >= 23
    i32x8 z8; for (int i = 0; i < 8; i++) z8[i] = 0;
    __builtin_amdgcn_tensor_load_to_lds(g0, g1, z4, z4, z8, 0);
#else
    __builtin_amdgcn_tensor_load_to_lds(g0, g1, z4, z4, 0);
#endif
}
#endif  // USE_TDM

// ---------------------------------------------------------------------------
// Conversion kernels (one-time, L2-resident): emit B operands pre-transposed
// (N-major) so GEMM staging is a pure row-contiguous copy -> TDM-friendly.
// ---------------------------------------------------------------------------
__global__ __launch_bounds__(256) void f32_to_bf16_kernel(const float* __restrict__ in,
                                                          bf* __restrict__ out, int n) {
    int i = blockIdx.x * 256 + threadIdx.x;
    if (i < n) out[i] = (bf)in[i];
}

// wcatT[c][k] = W(k, c) with W = [wq | wk | wv] columns; c in [0,3072), k in [0,2048)
__global__ __launch_bounds__(256) void build_wcatT_kernel(const float* __restrict__ wq,
                                                          const float* __restrict__ wk,
                                                          const float* __restrict__ wv,
                                                          bf* __restrict__ wcatT) {
    int i = blockIdx.x * 256 + threadIdx.x;   // over CW3*CD, k fastest (writes coalesced)
    if (i >= CW3 * CD) return;
    int c = i / CD;
    int k = i % CD;
    float v;
    if (c < 2048)      v = wq[(size_t)k * 2048 + c];
    else if (c < 2560) v = wk[(size_t)k * 512 + (c - 2048)];
    else               v = wv[(size_t)k * 512 + (c - 2560)];
    wcatT[i] = (bf)v;
}

// outT (cols x rows) = transpose(in (rows x cols)), f32 -> bf16
__global__ __launch_bounds__(256) void transpose_f32_bf16_kernel(const float* __restrict__ in,
                                                                 bf* __restrict__ outT,
                                                                 int rows, int cols) {
    int i = blockIdx.x * 256 + threadIdx.x;
    if (i >= rows * cols) return;
    int c = i / rows, r = i % rows;
    outT[i] = (bf)in[(size_t)r * cols + c];
}

// ---------------------------------------------------------------------------
// bf16 GEMM with pre-transposed B: C(f32, MxNc) = A(MxK) @ BT(NcxK)^T
// 128x128 tile, 256 threads (8 waves), double-buffered LDS, TDM staging.
// ---------------------------------------------------------------------------
__global__ __launch_bounds__(256) void gemm_bf16_kernel(const bf* __restrict__ A,
                                                        const bf* __restrict__ BT,
                                                        float* __restrict__ C,
                                                        int M, int Nc, int K) {
    __shared__ bf As[2][128 * 32];   // [row][k]
    __shared__ bf Bs[2][128 * 32];   // [col][k]

    const int tid  = threadIdx.x;
    const int lane = tid & 31;
    const int w    = tid >> 5;
    const int mL   = lane & 15;
    const int hi   = lane >> 4;

    const int nTilesN = Nc / 128;
    const int m0 = (blockIdx.x / nTilesN) * 128;
    const int n0 = (blockIdx.x % nTilesN) * 128;

    const bf* gA = A  + (size_t)m0 * K;
    const bf* gB = BT + (size_t)n0 * K;

    v8f acc[8];
    for (int nf = 0; nf < 8; nf++)
        for (int e = 0; e < 8; e++) acc[nf][e] = 0.0f;

    const int nk = K >> 5;

#if USE_TDM
    auto stage = [&](int buf, int k0) {
        if (tid < 32) {   // wave 0 drives the Tensor Data Mover (EXEC-independent)
            tdm_load_tile_128x32(gA + k0, lds_byte_offset(&As[buf][0]), (unsigned)K);
            tdm_load_tile_128x32(gB + k0, lds_byte_offset(&Bs[buf][0]), (unsigned)K);
        }
    };
    auto stage_wait = [&]() {
        if (tid < 32) __builtin_amdgcn_s_wait_tensorcnt(0);
    };
#else
    auto stage = [&](int buf, int k0) {
        // 512 uint4 per tile; each thread: 2 for A, 2 for B. All loads issued
        // before any LDS store so the compiler batches them (no per-elem waits).
        int i0 = tid, i1 = tid + 256;
        int r0 = i0 >> 2, c0 = i0 & 3;
        int r1 = i1 >> 2, c1 = i1 & 3;
        uint4 a0 = *(const uint4*)(gA + (size_t)r0 * K + k0 + c0 * 8);
        uint4 a1 = *(const uint4*)(gA + (size_t)r1 * K + k0 + c1 * 8);
        uint4 b0 = *(const uint4*)(gB + (size_t)r0 * K + k0 + c0 * 8);
        uint4 b1 = *(const uint4*)(gB + (size_t)r1 * K + k0 + c1 * 8);
        *(uint4*)&As[buf][r0 * 32 + c0 * 8] = a0;
        *(uint4*)&As[buf][r1 * 32 + c1 * 8] = a1;
        *(uint4*)&Bs[buf][r0 * 32 + c0 * 8] = b0;
        *(uint4*)&Bs[buf][r1 * 32 + c1 * 8] = b1;
    };
    auto stage_wait = [&]() {};
#endif

    stage(0, 0);
    stage_wait();
    __syncthreads();

    for (int kt = 0; kt < nk; kt++) {
        const int buf = kt & 1;
        if (kt + 1 < nk) stage(buf ^ 1, (kt + 1) * 32);
        if (kt + 2 < nk) {
            // L2 prefetch of the tile after next (global_prefetch_b8)
            __builtin_prefetch(gA + (size_t)(tid >> 1) * K + (kt + 2) * 32 + (tid & 1) * 16, 0, 1);
            __builtin_prefetch(gB + (size_t)(tid >> 1) * K + (kt + 2) * 32 + (tid & 1) * 16, 0, 1);
        }

        // ---- compute on buf: wave strip 16 rows x 128 cols, K-step 32
        Frag16 fa;
        {
            const unsigned* Au = (const unsigned*)&As[buf][0];
            int mrow = w * 16 + mL;
            for (int p = 0; p < 8; p++)
                fa.u[p] = Au[(mrow * 32 + a_frag_k(p, hi)) >> 1];
        }
        const unsigned* Bu = (const unsigned*)&Bs[buf][0];
        for (int nf = 0; nf < 8; nf++) {
            Frag16 fb;
            int ncol = nf * 16 + mL;
            for (int p = 0; p < 8; p++)
                fb.u[p] = Bu[(ncol * 32 + a_frag_k(p, hi)) >> 1];
            acc[nf] = __builtin_amdgcn_wmma_f32_16x16x32_bf16(
                false, fa.v, false, fb.v, (short)0, acc[nf], false, false);
        }

        stage_wait();
        __syncthreads();
    }

    for (int nf = 0; nf < 8; nf++) {
        int col = n0 + nf * 16 + mL;
        for (int i = 0; i < 8; i++) {
            int row = m0 + w * 16 + i + hi * 8;
            C[(size_t)row * Nc + col] = acc[nf][i];
        }
    }
}

// ---------------------------------------------------------------------------
// RoPE + head reductions (one thread per (b,n,d)):
//   Q1   (b,h,n,d) bf16 : rope(q)+1
//   Ksum (b,n,d)   bf16 : sum_h (rope(k)+1)
//   VsumT(b,d,n)   bf16 : sum_h v   (d-major for contiguous PV B-fragments)
// ---------------------------------------------------------------------------
__global__ __launch_bounds__(256) void rope_reduce_kernel(const float* __restrict__ qkv,
                                                          const float* __restrict__ cosb,
                                                          const float* __restrict__ sinb,
                                                          bf* __restrict__ Q1,
                                                          bf* __restrict__ Ksum,
                                                          bf* __restrict__ VsumT) {
    int i = blockIdx.x * 256 + threadIdx.x;    // over CB*CN*CHD = 524288
    if (i >= CB * CN * CHD) return;
    int d = i & 127;
    int n = (i >> 7) & (CN - 1);
    int b = i >> 18;

    float cv = cosb[n * CHD + d];
    float sv = sinb[n * CHD + d];
    int   d2  = (d < 64) ? d + 64 : d - 64;
    float sgn = (d < 64) ? -1.0f : 1.0f;

    const float* row = qkv + (size_t)(b * CN + n) * CW3;

    for (int h = 0; h < CQH; h++) {
        float qv = row[h * CHD + d];
        float qp = row[h * CHD + d2];
        float qr = qv * cv + sgn * qp * sv + 1.0f;
        Q1[(((size_t)(b * CQH + h)) * CN + n) * CHD + d] = (bf)qr;
    }
    float ks = 0.0f;
    for (int h = 0; h < CKVH; h++) {
        float kv = row[2048 + h * CHD + d];
        float kp = row[2048 + h * CHD + d2];
        ks += kv * cv + sgn * kp * sv + 1.0f;
    }
    Ksum[((size_t)b * CN + n) * CHD + d] = (bf)ks;

    float vs = 0.0f;
    for (int h = 0; h < CKVH; h++) vs += row[2560 + h * CHD + d];
    VsumT[((size_t)b * CHD + d) * CN + n] = (bf)vs;
}

// ---------------------------------------------------------------------------
// Flash attention over the head-summed KV. 8 waves/block; block = 128 query
// rows of one (b,h); each wave streams 32-wide key tiles for its 16-query tile.
// ---------------------------------------------------------------------------
__global__ __launch_bounds__(256) void flash_kernel(const bf* __restrict__ Q1,
                                                    const bf* __restrict__ Ksum,
                                                    const bf* __restrict__ VsumT,
                                                    bf* __restrict__ Oout) {
    __shared__ bf Plds[8 * 16 * 32];   // per-wave 16x32 P re-pack buffer

    const int tid  = threadIdx.x;
    const int lane = tid & 31;
    const int w    = tid >> 5;
    const int mL   = lane & 15;
    const int hi   = lane >> 4;

    const int qtiles = CN / 128;              // 16
    const int qt = blockIdx.x % qtiles;
    const int bh = blockIdx.x / qtiles;
    const int b  = bh >> 4;
    const int h  = bh & 15;

    const int q0 = qt * 128 + w * 16;

    Frag16 qa[4];   // Q A-fragments, K = 0..127, reused for every key tile
    {
        const unsigned* Qu = (const unsigned*)Q1;
        size_t rowbase = (((size_t)(b * CQH + h)) * CN + q0 + mL) * CHD;
        for (int kq = 0; kq < 4; kq++)
            for (int p = 0; p < 8; p++)
                qa[kq].u[p] = Qu[(rowbase + kq * 32 + a_frag_k(p, hi)) >> 1];
    }

    float Mrow[8], Lrow[8];
    v8f o[8];
    for (int i = 0; i < 8; i++) {
        Mrow[i] = -__builtin_inff();
        Lrow[i] = 0.0f;
        for (int e = 0; e < 8; e++) o[i][e] = 0.0f;
    }

    const float scale = 0.08838834764831845f;  // 1/sqrt(128)
    const int   qend  = q0 + 15;
    bf*       pbase  = Plds + w * 16 * 32;
    unsigned* pbaseu = (unsigned*)pbase;

    for (int s0 = 0; s0 <= qend; s0 += 32) {
        // scores: two 16x16 halves, contraction K=128 via 4 bf16 WMMA each
        v8f sc[2];
        const unsigned* Ku = (const unsigned*)Ksum;
        for (int half = 0; half < 2; half++) {
            v8f c;
            for (int e = 0; e < 8; e++) c[e] = 0.0f;
            size_t krow = ((size_t)b * CN + s0 + half * 16 + mL) * CHD;
            for (int kq = 0; kq < 4; kq++) {
                Frag16 fb;
                size_t kb = krow + kq * 32;
                for (int p = 0; p < 8; p++) fb.u[p] = Ku[(kb + a_frag_k(p, hi)) >> 1];
                c = __builtin_amdgcn_wmma_f32_16x16x32_bf16(
                    false, qa[kq].v, false, fb.v, (short)0, c, false, false);
            }
            sc[half] = c;
        }

        // mask + transform + online softmax (rows live across 16-lane halves)
        float alpha[8];
        for (int i = 0; i < 8; i++) {
            int gq  = q0 + i + hi * 8;
            int gs0 = s0 + mL;
            int gs1 = s0 + 16 + mL;
            float s0v = sc[0][i], s1v = sc[1][i];
            if (gs0 > gq || s0v == 128.0f) s0v = -__builtin_inff();
            else                           s0v = (s0v - 1.0f) * scale;
            if (gs1 > gq || s1v == 128.0f) s1v = -__builtin_inff();
            else                           s1v = (s1v - 1.0f) * scale;

            float mloc = fmaxf(s0v, s1v);
            for (int off = 8; off >= 1; off >>= 1)
                mloc = fmaxf(mloc, __shfl_xor(mloc, off, 16));
            float newM = fmaxf(Mrow[i], mloc);

            float a  = (Mrow[i] == -__builtin_inff()) ? 0.0f : __expf(Mrow[i] - newM);
            float p0 = (s0v == -__builtin_inff()) ? 0.0f : __expf(s0v - newM);
            float p1 = (s1v == -__builtin_inff()) ? 0.0f : __expf(s1v - newM);
            float rs = p0 + p1;
            for (int off = 8; off >= 1; off >>= 1)
                rs += __shfl_xor(rs, off, 16);

            Lrow[i]  = Lrow[i] * a + rs;
            Mrow[i]  = newM;
            alpha[i] = a;

            int m = i + hi * 8;
            pbase[m * 32 + mL]      = (bf)p0;
            pbase[m * 32 + 16 + mL] = (bf)p1;
        }

        for (int nf = 0; nf < 8; nf++)
            for (int i = 0; i < 8; i++) o[nf][i] *= alpha[i];

        // gather P as a 16x32 A-fragment (wave-internal; LDS ops are in-order)
        Frag16 pa;
        for (int p = 0; p < 8; p++)
            pa.u[p] = pbaseu[(mL * 32 + a_frag_k(p, hi)) >> 1];

        // O += P @ Vsum  (8 WMMA over the 128 head dims)
        const unsigned* Vu = (const unsigned*)VsumT;
        for (int nf = 0; nf < 8; nf++) {
            Frag16 fb;
            size_t vrow = ((size_t)b * CHD + nf * 16 + mL) * CN + s0;
            for (int p = 0; p < 8; p++) fb.u[p] = Vu[(vrow + a_frag_k(p, hi)) >> 1];
            o[nf] = __builtin_amdgcn_wmma_f32_16x16x32_bf16(
                false, pa.v, false, fb.v, (short)0, o[nf], false, false);
        }
    }

    // normalize + store O in (b, n, h*128+d) layout for the final GEMM
    for (int nf = 0; nf < 8; nf++) {
        int col = h * CHD + nf * 16 + mL;
        for (int i = 0; i < 8; i++) {
            float inv = (Lrow[i] > 0.0f) ? 1.0f / Lrow[i] : 0.0f;
            size_t row = (size_t)b * CN + q0 + i + hi * 8;
            Oout[row * (size_t)CD + col] = (bf)(o[nf][i] * inv);
        }
    }
}

// ---------------------------------------------------------------------------
// Host-side orchestration
// ---------------------------------------------------------------------------
extern "C" void kernel_launch(void* const* d_in, const int* in_sizes, int n_in,
                              void* d_out, int out_size, void* d_ws, size_t ws_size,
                              hipStream_t stream) {
    (void)in_sizes; (void)n_in; (void)out_size; (void)ws_size;

    const float* x    = (const float*)d_in[0];
    const float* wq   = (const float*)d_in[1];
    const float* wk   = (const float*)d_in[2];
    const float* wv   = (const float*)d_in[3];
    const float* wo   = (const float*)d_in[4];
    const float* cosb = (const float*)d_in[5];
    const float* sinb = (const float*)d_in[6];
    // d_in[7] = attn_mask: causal mask reproduced analytically, not read.
    float* out = (float*)d_out;

    char* ws = (char*)d_ws;
    bf*    xb    = (bf*)ws;    ws += (size_t)CB * CN * CD * 2;        // 16.8 MB
    bf*    wcatT = (bf*)ws;    ws += (size_t)CW3 * CD * 2;            // 12.6 MB
    bf*    wobT  = (bf*)ws;    ws += (size_t)CD * CD * 2;             //  8.4 MB
    float* qkv   = (float*)ws; ws += (size_t)CB * CN * CW3 * 4;       // 50.3 MB
    bf*    Q1    = (bf*)ws;    ws += (size_t)CB * CQH * CN * CHD * 2; // 16.8 MB
    bf*    Ksum  = (bf*)ws;    ws += (size_t)CB * CN * CHD * 2;       //  1.0 MB
    bf*    VsumT = (bf*)ws;    ws += (size_t)CB * CHD * CN * 2;       //  1.0 MB
    bf*    Oo    = (bf*)ws;    ws += (size_t)CB * CN * CD * 2;        // 16.8 MB

    const int nx  = CB * CN * CD;   // 8388608
    const int nwc = CW3 * CD;       // 6291456
    const int nwo = CD * CD;        // 4194304

    f32_to_bf16_kernel<<<nx / 256, 256, 0, stream>>>(x, xb, nx);
    build_wcatT_kernel<<<nwc / 256, 256, 0, stream>>>(wq, wk, wv, wcatT);
    transpose_f32_bf16_kernel<<<nwo / 256, 256, 0, stream>>>(wo, wobT, CD, CD);

    // qkv = xb @ wcatT^T   (4096 x 3072, K=2048)
    gemm_bf16_kernel<<<(CB * CN / 128) * (CW3 / 128), 256, 0, stream>>>(
        xb, wcatT, qkv, CB * CN, CW3, CD);

    rope_reduce_kernel<<<(CB * CN * CHD) / 256, 256, 0, stream>>>(
        qkv, cosb, sinb, Q1, Ksum, VsumT);

    // attention: one block per (b, h, 128-row q strip)
    flash_kernel<<<CB * CQH * (CN / 128), 256, 0, stream>>>(Q1, Ksum, VsumT, Oo);

    // out = Oo @ wobT^T   (4096 x 2048, K=2048)
    gemm_bf16_kernel<<<(CB * CN / 128) * (CD / 128), 256, 0, stream>>>(
        Oo, wobT, out, CB * CN, CD, CD);
}
